// HierarchicalCoarseGraining_45698452029584
// MI455X (gfx1250) — matrix operand order repaired and checked by reading, not compiled, not run on hardware
//
#include <hip/hip_runtime.h>

// Problem constants (match reference)
#define NN0 40000
#define EE0 1000000
#define HIDD 128
#define KK1 20000
#define KK2 10000
#define KK3 5000

typedef __attribute__((ext_vector_type(16))) _Float16 v16h;
typedef __attribute__((ext_vector_type(8)))  float    v8f;
typedef __attribute__((ext_vector_type(2)))  float    v2f;
typedef __attribute__((ext_vector_type(4)))  unsigned v4u;
typedef __attribute__((ext_vector_type(8)))  int      v8i;
typedef __attribute__((ext_vector_type(4)))  int      v4i;

__device__ __forceinline__ unsigned f2ord(float f){
  unsigned b = __float_as_uint(f);
  return (b & 0x80000000u) ? ~b : (b | 0x80000000u);
}
__device__ __forceinline__ float ord2f(unsigned u){
  return (u & 0x80000000u) ? __uint_as_float(u ^ 0x80000000u) : __uint_as_float(~u);
}
__device__ __forceinline__ float lrelu(float x){ return x > 0.f ? x : 0.2f*x; }

// ---------------------------------------------------------------- utilities
__global__ void fill_f32(float* p, float v, int n){
  int i = blockIdx.x*blockDim.x + threadIdx.x;
  if(i < n) p[i] = v;
}

__global__ void init_edges_k(const int* __restrict__ ei, int* __restrict__ src,
                             int* __restrict__ dst, float* __restrict__ mask, int E){
  int e = blockIdx.x*blockDim.x + threadIdx.x;
  if(e >= E) return;
  src[e] = ei[e];
  dst[e] = ei[E + e];
  mask[e] = 1.0f;
}

// ----------------------------------------------------- WMMA dense matmuls
// out[N,128] = A[N,3] @ W[3,128]  (K=3 padded to 4, f32 WMMA 16x16x4)
// Callers guarantee N % 16 == 0 and grid == N/16: no guards needed.
__global__ void mm_din_hid_wmma(const float* __restrict__ A,
                                const float* __restrict__ W,
                                float* __restrict__ out){
  int lane = threadIdx.x & 31;
  int wave = threadIdx.x >> 5;        // 0..7 : column tile
  int half = lane >> 4;               // 0: K=0,1 | 1: K=2,3(pad)
  int l15  = lane & 15;
  int row_base = blockIdx.x * 16;
  int arow = row_base + l15;

  v2f a;
  if(half == 0){
    a.x = A[arow*3 + 0];
    a.y = A[arow*3 + 1];
  } else {
    a.x = A[arow*3 + 2];
    a.y = 0.f;
  }
  int col = wave*16 + l15;
  v2f b;
  if(half == 0){
    b.x = W[0*HIDD + col];
    b.y = W[1*HIDD + col];
  } else {
    b.x = W[2*HIDD + col];
    b.y = 0.f;
  }
  v8f c = {};
  c = __builtin_amdgcn_wmma_f32_16x16x4_f32(false, a, false, b, (short)0, c, false, false);
  float* op = out + (size_t)(row_base + half*8)*HIDD + col;
#pragma unroll
  for(int i = 0; i < 8; i++) op[(size_t)i*HIDD] = c[i];
}

// out[N,3] = A[N,128] @ W[128,3]  (f16 WMMA 16x16x32, 4 K-steps, cols padded to 16)
// Validity is checked per 16-row strip, so it is wave-uniform; rows past N in a partial strip are not guarded.
__global__ void mm_hid_cg_wmma(const float* __restrict__ A,
                               const float* __restrict__ W,
                               float* __restrict__ out, int N){
  int lane = threadIdx.x & 31;
  int wave = threadIdx.x >> 5;        // 0..7 : row strip
  int half = lane >> 4;
  int l15  = lane & 15;
  int row_base = blockIdx.x*128 + wave*16;
  if(row_base >= N) return;           // wave-uniform exit
  const float* arowp = A + (size_t)(row_base + l15)*HIDD;
  int col = l15;

  v8f c = {};
#pragma unroll
  for(int s = 0; s < 4; s++){
    int kbase = s*32;
    float4 p0 = *(const float4*)(arowp + kbase + half*8);
    float4 p1 = *(const float4*)(arowp + kbase + half*8 + 4);
    float4 q0 = *(const float4*)(arowp + kbase + 16 + half*8);
    float4 q1 = *(const float4*)(arowp + kbase + 16 + half*8 + 4);
    v16h a;
    a[0]=(_Float16)p0.x; a[1]=(_Float16)p0.y; a[2]=(_Float16)p0.z; a[3]=(_Float16)p0.w;
    a[4]=(_Float16)p1.x; a[5]=(_Float16)p1.y; a[6]=(_Float16)p1.z; a[7]=(_Float16)p1.w;
    a[8]=(_Float16)q0.x; a[9]=(_Float16)q0.y; a[10]=(_Float16)q0.z; a[11]=(_Float16)q0.w;
    a[12]=(_Float16)q1.x; a[13]=(_Float16)q1.y; a[14]=(_Float16)q1.z; a[15]=(_Float16)q1.w;
    v16h b;
#pragma unroll
    for(int j = 0; j < 16; j++){
      int k = kbase + ((j < 8) ? (half*8 + j) : (16 + half*8 + (j - 8)));
      b[j] = (col < 3) ? (_Float16)W[k*3 + col] : (_Float16)0.f;
    }
    c = __builtin_amdgcn_wmma_f32_16x16x32_f16(false, a, false, b, (short)0, c, false, false);
  }
  if(col < 3){
    float* op = out + (size_t)(row_base + half*8)*3 + col;
#pragma unroll
    for(int i = 0; i < 8; i++) op[(size_t)i*3] = c[i];
  }
}

// ----------------------------------------------------------------- GCN
__global__ void deg_scatter_k(const int* __restrict__ dst, const float* __restrict__ mask,
                              float* __restrict__ deg, int E){
  int e = blockIdx.x*blockDim.x + threadIdx.x;
  if(e >= E) return;
  float m = mask[e];
  if(m > 0.f) atomicAdd(&deg[dst[e]], m);
}

__global__ void finish_dis_k(float* __restrict__ dis, int N){
  int n = blockIdx.x*blockDim.x + threadIdx.x;
  if(n < N) dis[n] = rsqrtf(dis[n] + 1.0f);
}

// out[n,c] = dis[n]^2 * h[n,c] + b[c]
__global__ void gcn_init_k(const float* __restrict__ dis, const float* __restrict__ h,
                           const float* __restrict__ b, float* __restrict__ out,
                           int N, int C){
  long long idx = (long long)blockIdx.x*blockDim.x + threadIdx.x;
  if(idx >= (long long)N*C) return;
  int n = (int)(idx / C);
  int c = (int)(idx % C);
  float d = dis[n];
  out[idx] = d*d*h[idx] + b[c];
}

// per (edge, 4-float chunk of 128)
__global__ void gcn_scatter128_k(const int* __restrict__ src, const int* __restrict__ dst,
                                 const float* __restrict__ mask, const float* __restrict__ dis,
                                 const float* __restrict__ h, float* __restrict__ out, int E){
  long long idx = (long long)blockIdx.x*blockDim.x + threadIdx.x;
  int e = (int)(idx >> 5);
  if(e >= E) return;
  float m = mask[e];
  if(m <= 0.f) return;
  int s = src[e], d = dst[e];
  float coef = dis[s]*dis[d]*m;
  int c = (int)(idx & 31) * 4;
  const float4 hv = *(const float4*)(h + (size_t)s*HIDD + c);
  float* op = out + (size_t)d*HIDD + c;
  atomicAdd(op + 0, coef*hv.x);
  atomicAdd(op + 1, coef*hv.y);
  atomicAdd(op + 2, coef*hv.z);
  atomicAdd(op + 3, coef*hv.w);
}

__global__ void gcn_scatter3_k(const int* __restrict__ src, const int* __restrict__ dst,
                               const float* __restrict__ mask, const float* __restrict__ dis,
                               const float* __restrict__ h, float* __restrict__ out, int E){
  int e = blockIdx.x*blockDim.x + threadIdx.x;
  if(e >= E) return;
  float m = mask[e];
  if(m <= 0.f) return;
  int s = src[e], d = dst[e];
  float coef = dis[s]*dis[d]*m;
  atomicAdd(&out[(size_t)d*3 + 0], coef*h[(size_t)s*3 + 0]);
  atomicAdd(&out[(size_t)d*3 + 1], coef*h[(size_t)s*3 + 1]);
  atomicAdd(&out[(size_t)d*3 + 2], coef*h[(size_t)s*3 + 2]);
}

// ----------------------------------------------------------------- TopK pool
__global__ void pw_norm_k(const float* __restrict__ pw, float* __restrict__ invn){
  __shared__ float red[128];
  int t = threadIdx.x;
  float v = pw[t];
  red[t] = v*v;
  __syncthreads();
  for(int s = 64; s > 0; s >>= 1){
    if(t < s) red[t] += red[t + s];
    __syncthreads();
  }
  if(t == 0) invn[0] = rsqrtf(red[0]);
}

__global__ void score_k(const float* __restrict__ h, const float* __restrict__ pw,
                        const float* __restrict__ invn, float* __restrict__ score, int N){
  int n = blockIdx.x*blockDim.x + threadIdx.x;
  if(n >= N) return;
  const float* hp = h + (size_t)n*HIDD;
  float s = 0.f;
#pragma unroll 8
  for(int c = 0; c < HIDD; c += 4){
    float4 hv = *(const float4*)(hp + c);
    float4 pv = *(const float4*)(pw + c);
    s += hv.x*pv.x + hv.y*pv.y + hv.z*pv.z + hv.w*pv.w;
  }
  score[n] = tanhf(s * invn[0]);
}

// Issue a Tensor-Data-Mover load of `lim` floats from `gptr` into LDS at
// byte offset `lds_off`, then drain TENSORcnt. One wave per workgroup.
// D# layout per CDNA5 ISA.
__device__ __forceinline__ void tdm_load_row_to_lds(const float* gptr, unsigned lds_off, int lim){
  unsigned long long ga = (unsigned long long)(const void*)gptr;
  v4u g0;
  g0[0] = 1u;                                           // count=1 (valid user D#)
  g0[1] = lds_off;                                      // lds_addr (bytes)
  g0[2] = (unsigned)(ga & 0xffffffffu);                 // global_addr[31:0]
  g0[3] = (unsigned)((ga >> 32) & 0x1ffffffu) | (2u << 30); // global_addr[56:32], type=2
  unsigned ulim = (unsigned)lim;
  v8i g1;
  g1[0] = 0x00020000;                                   // data_size=4B, no flags
  g1[1] = (int)(ulim << 16);                            // tensor_dim0[15:0]
  g1[2] = (int)((ulim >> 16) | (1u << 16));             // tensor_dim0[31:16], tensor_dim1=1
  g1[3] = (int)(ulim << 16);                            // tile_dim0 = lim
  g1[4] = 1;                                            // tile_dim1=1, tile_dim2=0
  g1[5] = (int)ulim;                                    // tensor_dim0_stride[31:0]
  g1[6] = 0;
  g1[7] = 0;
  v4i gz = {};
#if defined(__clang_major__) && (__clang_major__ >= 23)
  v8i g4 = {};
  __builtin_amdgcn_tensor_load_to_lds(g0, g1, gz, gz, g4, 0);
#else
  __builtin_amdgcn_tensor_load_to_lds(g0, g1, gz, gz, 0);
#endif
  __builtin_amdgcn_s_wait_tensorcnt(0);
}

// rank[n] = #{j : score[j] > score[n] or (== and j < n)}  (exact PyG top_k order)
// Score chunks are staged into LDS by the TDM (wave 0 issues, all waves consume).
__global__ void rank_k(const float* __restrict__ score, int* __restrict__ rank, int N){
  __shared__ float sc[256];
  int n = blockIdx.x*256 + threadIdx.x;
  float my = (n < N) ? score[n] : 0.f;
  unsigned lds_off = (unsigned)(unsigned long long)(void*)&sc[0];
  int r = 0;
  for(int base = 0; base < N; base += 256){
    int lim = min(256, N - base);
    __syncthreads();
    if(threadIdx.x < 32){                 // wave 0 only: async tensor load
      tdm_load_row_to_lds(score + base, lds_off, lim);
    }
    __syncthreads();
    for(int t = 0; t < lim; t++){
      float v = sc[t];
      int jj = base + t;
      r += (v > my) || (v == my && jj < n);
    }
  }
  if(n < N) rank[n] = r;
}

// hp[rank[n],:] = h[n,:]*score[n] for kept; newidx
__global__ void compact_k(const float* __restrict__ h, const float* __restrict__ score,
                          const int* __restrict__ rank, float* __restrict__ hp,
                          int* __restrict__ newidx, int N, int K){
  long long idx = (long long)blockIdx.x*blockDim.x + threadIdx.x;
  int n = (int)(idx >> 5);
  if(n >= N) return;
  int c = (int)(idx & 31) * 4;
  int r = rank[n];
  if(r < K){
    if(c == 0) newidx[n] = r;
    float sv = score[n];
    float4 hv = *(const float4*)(h + (size_t)n*HIDD + c);
    float4 o = make_float4(hv.x*sv, hv.y*sv, hv.z*sv, hv.w*sv);
    *(float4*)(hp + (size_t)r*HIDD + c) = o;
  } else {
    if(c == 0) newidx[n] = -1;
  }
}

__global__ void edge_relabel_k(const int* __restrict__ srcI, const int* __restrict__ dstI,
                               const float* __restrict__ maskI, const int* __restrict__ newidx,
                               int* __restrict__ srcO, int* __restrict__ dstO,
                               float* __restrict__ maskO, int E){
  int e = blockIdx.x*blockDim.x + threadIdx.x;
  if(e >= E) return;
  int ns = newidx[srcI[e]];
  int nd = newidx[dstI[e]];
  float m = maskI[e] * ((ns >= 0) ? 1.f : 0.f) * ((nd >= 0) ? 1.f : 0.f);
  srcO[e] = ns > 0 ? ns : 0;
  dstO[e] = nd > 0 ? nd : 0;
  maskO[e] = m;
}

// ----------------------------------------------------------------- GAT
__global__ void gat_node_k(const float* __restrict__ h, const float* __restrict__ asrc,
                           const float* __restrict__ adst, float* __restrict__ sarr,
                           float* __restrict__ darr, float* __restrict__ eself,
                           unsigned* __restrict__ mord, int N){
  int n = blockIdx.x*blockDim.x + threadIdx.x;
  if(n >= N) return;
  const float* hp = h + (size_t)n*HIDD;
  float s = 0.f, d = 0.f;
#pragma unroll 8
  for(int c = 0; c < HIDD; c += 4){
    float4 hv = *(const float4*)(hp + c);
    float4 as = *(const float4*)(asrc + c);
    float4 ad = *(const float4*)(adst + c);
    s += hv.x*as.x + hv.y*as.y + hv.z*as.z + hv.w*as.w;
    d += hv.x*ad.x + hv.y*ad.y + hv.z*ad.z + hv.w*ad.w;
  }
  sarr[n] = s;
  darr[n] = d;
  float e = lrelu(s + d);
  eself[n] = e;
  mord[n] = f2ord(e);
}

__global__ void gat_edge_max_k(const int* __restrict__ src, const int* __restrict__ dst,
                               const float* __restrict__ mask, const float* __restrict__ sarr,
                               const float* __restrict__ darr, unsigned* __restrict__ mord, int E){
  int e = blockIdx.x*blockDim.x + threadIdx.x;
  if(e >= E) return;
  if(mask[e] <= 0.f) return;
  float ev = lrelu(sarr[src[e]] + darr[dst[e]]);
  atomicMax(&mord[dst[e]], f2ord(ev));
}

__global__ void gat_init_k(const float* __restrict__ h, const float* __restrict__ eself,
                           const unsigned* __restrict__ mord, float* __restrict__ denom,
                           float* __restrict__ num, int N){
  long long idx = (long long)blockIdx.x*blockDim.x + threadIdx.x;
  int n = (int)(idx >> 5);
  if(n >= N) return;
  int c = (int)(idx & 31) * 4;
  float m = ord2f(mord[n]);
  float ex = __expf(eself[n] - m);
  if(c == 0) denom[n] = ex;
  float4 hv = *(const float4*)(h + (size_t)n*HIDD + c);
  float4 o = make_float4(ex*hv.x, ex*hv.y, ex*hv.z, ex*hv.w);
  *(float4*)(num + (size_t)n*HIDD + c) = o;
}

__global__ void gat_edge_accum_k(const int* __restrict__ src, const int* __restrict__ dst,
                                 const float* __restrict__ mask, const float* __restrict__ sarr,
                                 const float* __restrict__ darr, const unsigned* __restrict__ mord,
                                 const float* __restrict__ h, float* __restrict__ denom,
                                 float* __restrict__ num, int E){
  long long idx = (long long)blockIdx.x*blockDim.x + threadIdx.x;
  int e = (int)(idx >> 5);
  if(e >= E) return;
  if(mask[e] <= 0.f) return;
  int s = src[e], d = dst[e];
  float ev = lrelu(sarr[s] + darr[d]);
  float ex = __expf(ev - ord2f(mord[d]));
  int c = (int)(idx & 31) * 4;
  if(c == 0) atomicAdd(&denom[d], ex);
  float4 hv = *(const float4*)(h + (size_t)s*HIDD + c);
  float* np = num + (size_t)d*HIDD + c;
  atomicAdd(np + 0, ex*hv.x);
  atomicAdd(np + 1, ex*hv.y);
  atomicAdd(np + 2, ex*hv.z);
  atomicAdd(np + 3, ex*hv.w);
}

__global__ void gat_final_k(const float* __restrict__ num, const float* __restrict__ denom,
                            const float* __restrict__ b, float* __restrict__ out, int N){
  long long idx = (long long)blockIdx.x*blockDim.x + threadIdx.x;
  if(idx >= (long long)N*HIDD) return;
  int n = (int)(idx >> 7);
  int c = (int)(idx & 127);
  out[idx] = num[idx] / denom[n] + b[c];
}

// ----------------------------------------------------------------- host side
static inline unsigned gup(long long n, int b){ return (unsigned)((n + b - 1) / b); }

struct Graph { const int* src; const int* dst; const float* mask; };

static void compute_dis(const Graph& g, float* dis, int N, hipStream_t st){
  fill_f32<<<gup(N,256),256,0,st>>>(dis, 0.f, N);
  deg_scatter_k<<<gup(EE0,256),256,0,st>>>(g.dst, g.mask, dis, EE0);
  finish_dis_k<<<gup(N,256),256,0,st>>>(dis, N);
}

static void run_decoder_gcn(const float* hp, int N, const Graph& g,
                            const float* Wd, const float* bd,
                            float* hw, float* dis, float* out, hipStream_t st){
  compute_dis(g, dis, N, st);
  mm_hid_cg_wmma<<<gup(N,128),256,0,st>>>(hp, Wd, hw, N);
  gcn_init_k<<<gup((long long)N*3,256),256,0,st>>>(dis, hw, bd, out, N, 3);
  gcn_scatter3_k<<<gup(EE0,256),256,0,st>>>(g.src, g.dst, g.mask, dis, hw, out, EE0);
}

static void run_gat(const float* xin, int N, const Graph& g,
                    const float* W, const float* asrc, const float* adst, const float* b,
                    float* h, float* sarr, float* darr, float* eself, unsigned* mord,
                    float* denom, float* num, float* out, hipStream_t st){
  mm_din_hid_wmma<<<gup(N,16),256,0,st>>>(xin, W, h);
  gat_node_k<<<gup(N,256),256,0,st>>>(h, asrc, adst, sarr, darr, eself, mord, N);
  gat_edge_max_k<<<gup(EE0,256),256,0,st>>>(g.src, g.dst, g.mask, sarr, darr, mord, EE0);
  gat_init_k<<<gup((long long)N*32,256),256,0,st>>>(h, eself, mord, denom, num, N);
  gat_edge_accum_k<<<gup((long long)EE0*32,256),256,0,st>>>(g.src, g.dst, g.mask, sarr, darr,
                                                           mord, h, denom, num, EE0);
  gat_final_k<<<gup((long long)N*HIDD,256),256,0,st>>>(num, denom, b, out, N);
}

static void run_pool(const float* hin, int N, int K, const float* pw,
                     const Graph& gi, int* srcO, int* dstO, float* maskO,
                     float* hp, float* score, int* rank, int* newidx, float* invn,
                     hipStream_t st){
  pw_norm_k<<<1,128,0,st>>>(pw, invn);
  score_k<<<gup(N,256),256,0,st>>>(hin, pw, invn, score, N);
  rank_k<<<gup(N,256),256,0,st>>>(score, rank, N);
  compact_k<<<gup((long long)N*32,256),256,0,st>>>(hin, score, rank, hp, newidx, N, K);
  edge_relabel_k<<<gup(EE0,256),256,0,st>>>(gi.src, gi.dst, gi.mask, newidx,
                                           srcO, dstO, maskO, EE0);
}

extern "C" void kernel_launch(void* const* d_in, const int* in_sizes, int n_in,
                              void* d_out, int out_size, void* d_ws, size_t ws_size,
                              hipStream_t stream){
  const float* x       = (const float*)d_in[0];
  const int*   ei      = (const int*)  d_in[1];
  const float* W_enc0  = (const float*)d_in[2];
  const float* b_enc0  = (const float*)d_in[3];
  const float* W_gat1  = (const float*)d_in[4];
  const float* att_s1  = (const float*)d_in[5];
  const float* att_d1  = (const float*)d_in[6];
  const float* b_gat1  = (const float*)d_in[7];
  const float* W_gat2  = (const float*)d_in[8];
  const float* att_s2  = (const float*)d_in[9];
  const float* att_d2  = (const float*)d_in[10];
  const float* b_gat2  = (const float*)d_in[11];
  const float* Wd0     = (const float*)d_in[12];
  const float* bd0     = (const float*)d_in[13];
  const float* Wd1     = (const float*)d_in[14];
  const float* bd1     = (const float*)d_in[15];
  const float* Wd2     = (const float*)d_in[16];
  const float* bd2     = (const float*)d_in[17];
  const float* pw0     = (const float*)d_in[18];
  const float* pw1     = (const float*)d_in[19];
  const float* pw2     = (const float*)d_in[20];
  float* out = (float*)d_out;

  // ---- workspace carve-up
  char* w = (char*)d_ws;
  auto take = [&](size_t bytes){ char* p = w; w += (bytes + 15) & ~(size_t)15; return p; };
  float*    h0    = (float*)   take((size_t)NN0*HIDD*4);  // xW / GAT h
  float*    g0    = (float*)   take((size_t)NN0*HIDD*4);  // conv output (pool input)
  float*    hp    = (float*)   take((size_t)KK1*HIDD*4);  // pooled features / GAT num
  float*    hw    = (float*)   take((size_t)KK1*4*4);     // decoder xW [K,3]
  float*    dis   = (float*)   take((size_t)NN0*4);
  float*    score = (float*)   take((size_t)NN0*4);
  int*      rank  = (int*)     take((size_t)NN0*4);
  int*      nidx  = (int*)     take((size_t)NN0*4);
  float*    invn  = (float*)   take(16);
  float*    sarr  = (float*)   take((size_t)KK1*4);
  float*    darr  = (float*)   take((size_t)KK1*4);
  float*    eself = (float*)   take((size_t)KK1*4);
  unsigned* mord  = (unsigned*)take((size_t)KK1*4);
  float*    denom = (float*)   take((size_t)KK1*4);
  int*      srcA  = (int*)     take((size_t)EE0*4);
  int*      dstA  = (int*)     take((size_t)EE0*4);
  float*    maskA = (float*)   take((size_t)EE0*4);
  int*      srcB  = (int*)     take((size_t)EE0*4);
  int*      dstB  = (int*)     take((size_t)EE0*4);
  float*    maskB = (float*)   take((size_t)EE0*4);

  Graph gA{srcA, dstA, maskA};
  Graph gB{srcB, dstB, maskB};

  // ---- level 0: GCN encoder on full graph
  init_edges_k<<<gup(EE0,256),256,0,stream>>>(ei, srcA, dstA, maskA, EE0);
  mm_din_hid_wmma<<<gup(NN0,16),256,0,stream>>>(x, W_enc0, h0);
  compute_dis(gA, dis, NN0, stream);
  gcn_init_k<<<gup((long long)NN0*HIDD,256),256,0,stream>>>(dis, h0, b_enc0, g0, NN0, HIDD);
  gcn_scatter128_k<<<gup((long long)EE0*32,256),256,0,stream>>>(srcA, dstA, maskA, dis, h0, g0, EE0);

  // pool0: graph A -> B
  run_pool(g0, NN0, KK1, pw0, gA, srcB, dstB, maskB, hp, score, rank, nidx, invn, stream);
  // decoder 0 -> out[0 : K1*3)
  run_decoder_gcn(hp, KK1, gB, Wd0, bd0, hw, dis, out, stream);

  // ---- level 1: GAT on graph B
  run_gat(out, KK1, gB, W_gat1, att_s1, att_d1, b_gat1,
          h0, sarr, darr, eself, mord, denom, hp, g0, stream);
  run_pool(g0, KK1, KK2, pw1, gB, srcA, dstA, maskA, hp, score, rank, nidx, invn, stream);
  run_decoder_gcn(hp, KK2, gA, Wd1, bd1, hw, dis, out + (size_t)KK1*3, stream);

  // ---- level 2: GAT on graph A
  run_gat(out + (size_t)KK1*3, KK2, gA, W_gat2, att_s2, att_d2, b_gat2,
          h0, sarr, darr, eself, mord, denom, hp, g0, stream);
  run_pool(g0, KK2, KK3, pw2, gA, srcB, dstB, maskB, hp, score, rank, nidx, invn, stream);
  run_decoder_gcn(hp, KK3, gB, Wd2, bd2, hw, dis, out + (size_t)(KK1+KK2)*3, stream);

  (void)in_sizes; (void)n_in; (void)out_size; (void)ws_size;
}
